// SRM_Root_25795573580331
// MI455X (gfx1250) — compile-verified, matrix operand order checked by reading, not run on hardware
//
#include <hip/hip_runtime.h>
#include <hip/hip_bf16.h>

typedef __attribute__((ext_vector_type(16))) _Float16 v16h;
typedef __attribute__((ext_vector_type(8)))  float    v8f;

// ---------------------------------------------------------------------------
// Kernel 1: syn = S @ C.T via v_wmma_f32_16x16x32_f16, one wave per 16 rows.
// Branchless inner loop: unconditional clamped float2 loads, tail masked to
// zero on the B side only (zeros kill any clamped-garbage A values).
// Spikes are {0,1} and C is {0,1}, so f16 is exact.
// ---------------------------------------------------------------------------
__global__ __launch_bounds__(32) void synapse_gemm_wmma(
    const float* __restrict__ S,   // [T, E] row-major
    const float* __restrict__ C,   // [16, E] row-major (C.T is B)
    float* __restrict__ out,       // [T, 16]
    int T, int E)
{
    const int lane = threadIdx.x & 31;
    const int n    = lane & 15;                  // A row (M) == B/D column (N)
    const int hi   = lane >> 4;                  // half-wave select
    const int t0   = blockIdx.x * 16;

    const int rowA = (t0 + n < T) ? (t0 + n) : (T - 1);   // clamped A row
    const float* __restrict__ Arow = S + (size_t)rowA * E;
    const float* __restrict__ Brow = C + (size_t)n * E;

    v8f acc = {};

    #pragma unroll 2
    for (int k0 = 0; k0 < E; k0 += 32) {
        // prefetch next 128B block of this lane's A row (clamped, speculative)
        {
            int pc = k0 + 32 + 16 * hi;
            pc = (pc < E) ? pc : (E - 1);
            __builtin_prefetch(Arow + pc, 0, 1);
        }

        v16h a, b;
        #pragma unroll
        for (int vv = 0; vv < 8; ++vv) {
            // 16-bit A 16x32 layout: K pair base = (vv<4?0:16) + 2*(vv&3) + 8*hi
            const int kb = ((vv < 4) ? 0 : 16) + 2 * (vv & 3) + 8 * hi;
            const int c0 = k0 + kb;                       // even; E even
            const bool ok = (c0 < E);
            const int  ca = ok ? c0 : 0;                  // clamped address

            const float2 av = *(const float2*)(Arow + ca);
            const float2 bv = *(const float2*)(Brow + ca);

            a[2 * vv]     = (_Float16)av.x;               // garbage*0 in tail
            a[2 * vv + 1] = (_Float16)av.y;
            b[2 * vv]     = (_Float16)(ok ? bv.x : 0.0f); // exact zero tail
            b[2 * vv + 1] = (_Float16)(ok ? bv.y : 0.0f);
        }

        acc = __builtin_amdgcn_wmma_f32_16x16x32_f16(
                  false, a, false, b, (short)0, acc, false, false);
    }

    // C/D layout: VGPR r -> M = r + 8*hi, N = lane & 15
    #pragma unroll
    for (int r = 0; r < 8; ++r) {
        const int row = t0 + 8 * hi + r;
        if (row < T) out[(size_t)row * 16 + n] = acc[r];
    }
}

// ---------------------------------------------------------------------------
// Kernel 2: e_kern/i_kern (alpha kernels) + hist_kern -> out_filters block of
// d_out ([33,200] = vstack(e, i, hist)) and workspace copies.
// ---------------------------------------------------------------------------
__global__ __launch_bounds__(256) void build_filters(
    const float* __restrict__ Tau_e, const float* __restrict__ Tau_i,
    const float* __restrict__ W_e,   const float* __restrict__ W_i,
    const float* __restrict__ Del_e, const float* __restrict__ Del_i,
    const float* __restrict__ W_hist,const float* __restrict__ cos_basis,
    float* __restrict__ out_filters,
    float* __restrict__ ws_ek, float* __restrict__ ws_ik, float* __restrict__ ws_h)
{
    const int idx = blockIdx.x * blockDim.x + threadIdx.x;
    if (idx < 3200) {
        const int s = idx / 200;
        const int t = idx % 200;
        const float tf = (float)t;

        float te = fmaxf(tf - __expf(Del_e[s]), 0.0f) / __expf(Tau_e[s]);
        float ev = te * __expf(-te) * __expf(W_e[s]);
        out_filters[idx] = ev;
        ws_ek[idx]       = ev;

        float ti = fmaxf(tf - __expf(Del_i[s]), 0.0f) / __expf(Tau_i[s]);
        float iv = -(ti * __expf(-ti) * __expf(W_i[s]));
        out_filters[3200 + idx] = iv;
        ws_ik[idx]              = iv;
    } else if (idx < 3400) {
        const int t = idx - 3200;
        float h = 0.0f;
        #pragma unroll
        for (int b = 0; b < 16; ++b) h += W_hist[b] * cos_basis[b * 200 + t];
        out_filters[6400 + t] = h;
        ws_h[t]               = h;
    }
}

// ---------------------------------------------------------------------------
// Kernel 3: causal depthwise filter (kernels resident in LDS) + binary-tree
// tanh pass -> drive[t].
// ---------------------------------------------------------------------------
__global__ __launch_bounds__(256) void filt_tree(
    const float* __restrict__ synE, const float* __restrict__ synI,
    const float* __restrict__ eK_g, const float* __restrict__ iK_g,
    const int*   __restrict__ Cden_g,
    const float* __restrict__ Theta_g, const float* __restrict__ Wsub_g,
    float* __restrict__ drive_out, int T)
{
    __shared__ float eK[3200], iK[3200];
    __shared__ float theta[16], ews[16];
    __shared__ int   cden[256];

    for (int i = threadIdx.x; i < 3200; i += 256) { eK[i] = eK_g[i]; iK[i] = iK_g[i]; }
    if (threadIdx.x < 256) cden[threadIdx.x] = Cden_g[threadIdx.x];
    if (threadIdx.x < 16) {
        theta[threadIdx.x] = Theta_g[threadIdx.x];
        ews[threadIdx.x]   = __expf(Wsub_g[threadIdx.x]);
    }
    __syncthreads();

    const int t = blockIdx.x * 256 + threadIdx.x;
    if (t >= T) return;

    float acc[16];
    #pragma unroll
    for (int s = 0; s < 16; ++s) acc[s] = 0.0f;

    const int lim = (t < 200) ? t : 200;          // j < min(t, 200)
    for (int j = 0; j < lim; ++j) {
        const float* re = synE + (size_t)(t - 1 - j) * 16;
        const float* ri = synI + (size_t)(t - 1 - j) * 16;
        #pragma unroll
        for (int s = 0; s < 16; ++s)
            acc[s] += re[s] * eK[s * 200 + j] + ri[s] * iK[s * 200 + j];
    }

    float col[16];
    for (int idx = 15; idx >= 1; --idx) {
        float v = acc[idx] + theta[idx];
        #pragma unroll
        for (int j = 0; j < 16; ++j)
            if (cden[idx * 16 + j] == 1) v += col[j] * ews[j];
        col[idx] = tanhf(v);
    }
    float drive = acc[0] + theta[0];
    #pragma unroll
    for (int j = 0; j < 16; ++j)
        if (cden[j] == 1) drive += col[j] * ews[j];

    drive_out[t] = drive;
}

// ---------------------------------------------------------------------------
// Kernel 4: serial spike scan, one wave32; LDS ring of last 256 spikes;
// feedback[t] = sum_{d=1..200} spk[t-d]*hist[d-1]; shuffle reduction.
// ---------------------------------------------------------------------------
__global__ __launch_bounds__(32) void scan_spikes(
    const float* __restrict__ drive, const float* __restrict__ hist,
    float* __restrict__ spk_out, int T)
{
    __shared__ float ring[256];
    const int lane = threadIdx.x;
    for (int i = lane; i < 256; i += 32) ring[i] = 0.0f;

    float ht[7];
    #pragma unroll
    for (int i = 0; i < 7; ++i) {
        const int d = lane + 1 + 32 * i;
        ht[i] = (d <= 200) ? hist[d - 1] : 0.0f;
    }
    __syncthreads();

    for (int t = 0; t < T; ++t) {
        float partial = 0.0f;
        #pragma unroll
        for (int i = 0; i < 7; ++i) {
            const int d = lane + 1 + 32 * i;
            const float s = ring[(t - d) & 255];
            partial += ((d <= t) ? s : 0.0f) * ht[i];
        }
        #pragma unroll
        for (int off = 16; off >= 1; off >>= 1)
            partial += __shfl_xor(partial, off, 32);

        const float val = partial + drive[t];
        const float spk = (val > 0.0f) ? 1.0f : 0.0f;   // heaviside(x, 0)
        if (lane == 0) { ring[t & 255] = spk; spk_out[t] = spk; }
        __syncthreads();
    }
}

// ---------------------------------------------------------------------------
extern "C" void kernel_launch(void* const* d_in, const int* in_sizes, int n_in,
                              void* d_out, int out_size, void* d_ws, size_t ws_size,
                              hipStream_t stream) {
    const float* S_e    = (const float*)d_in[0];
    const float* S_i    = (const float*)d_in[1];
    const int*   C_den  = (const int*)  d_in[2];
    const float* C_se   = (const float*)d_in[3];
    const float* C_si   = (const float*)d_in[4];
    const float* cosb   = (const float*)d_in[5];
    const float* Tau_e  = (const float*)d_in[6];
    const float* Tau_i  = (const float*)d_in[7];
    const float* W_e    = (const float*)d_in[8];
    const float* W_i    = (const float*)d_in[9];
    const float* Del_e  = (const float*)d_in[10];
    const float* Del_i  = (const float*)d_in[11];
    const float* W_sub  = (const float*)d_in[12];
    const float* W_hist = (const float*)d_in[13];
    const float* Theta  = (const float*)d_in[14];

    const int E_e = in_sizes[3] / 16;            // 2000
    const int E_i = in_sizes[4] / 16;            // 500
    const int T   = in_sizes[0] / E_e;           // 20000

    float* out = (float*)d_out;
    float* ws  = (float*)d_ws;
    float* synE    = ws;                         // T*16
    float* synI    = ws + (size_t)T * 16;        // T*16
    float* wsEK    = synI + (size_t)T * 16;      // 3200
    float* wsIK    = wsEK + 3200;                // 3200
    float* wsH     = wsIK + 3200;                // 200
    float* wsDrive = wsH + 200;                  // T

    const int tiles = (T + 15) / 16;
    synapse_gemm_wmma<<<tiles, 32, 0, stream>>>(S_e, C_se, synE, T, E_e);
    synapse_gemm_wmma<<<tiles, 32, 0, stream>>>(S_i, C_si, synI, T, E_i);
    build_filters<<<14, 256, 0, stream>>>(Tau_e, Tau_i, W_e, W_i, Del_e, Del_i,
                                          W_hist, cosb, out + T, wsEK, wsIK, wsH);
    filt_tree<<<(T + 255) / 256, 256, 0, stream>>>(synE, synI, wsEK, wsIK,
                                                   C_den, Theta, W_sub, wsDrive, T);
    scan_spikes<<<1, 32, 0, stream>>>(wsDrive, wsH, out, T);
}
